// char_rnn_lstm_27831388078750
// MI455X (gfx1250) — compile-verified, matrix operand order checked by reading, not run on hardware
//
#include <hip/hip_runtime.h>
#include <hip/hip_bf16.h>
#include <stdint.h>

typedef __bf16 bf16_t;
typedef bf16_t v16bf __attribute__((ext_vector_type(16)));
typedef bf16_t v8bf  __attribute__((ext_vector_type(8)));
typedef float  v8f   __attribute__((ext_vector_type(8)));

#define B_   128
#define T_   512
#define V_   96
#define H_   512
#define G4H  2048
#define TC_  128          // timestep chunk
#define NWG_SCAN 16       // scan workgroups: H/32

// ---------- helpers ----------

__device__ __forceinline__ bf16_t f2bf(float x) { return (bf16_t)x; }

__device__ __forceinline__ float sigf(float x) { return 1.0f / (1.0f + __expf(-x)); }

// Load one 16x32 bf16 WMMA operand tile (A-layout; B uses same pattern on W rows,
// since B[k][n] = W[n][k] and column n's K-values live in lane n).
// ISA 7.12.2: lane L (q=L/16, r=L%16) holds K = [k+q*8 .. +8) and [k+16+q*8 .. +8).
__device__ __forceinline__ v16bf load_frag(const bf16_t* __restrict__ base,
                                           int ld, int row0, int k0, int lane) {
  int r = row0 + (lane & 15);
  int q = lane >> 4;
  const bf16_t* p = base + (size_t)r * ld + k0 + q * 8;
  v8bf lo = *(const v8bf*)(p);
  v8bf hi = *(const v8bf*)(p + 16);
  v16bf out;
#pragma unroll
  for (int i = 0; i < 8; ++i) { out[i] = lo[i]; out[i + 8] = hi[i]; }
  return out;
}

#define WMMA_BF16(a, b, c) \
  __builtin_amdgcn_wmma_f32_16x16x32_bf16(false, (a), false, (b), (short)0, (c), false, false)

// ---------- conversion / init kernels ----------

// fp32 [M,Kin] -> bf16 [M, ldd] with zero pad to Kout
__global__ void cvt_pad_kernel(const float* __restrict__ src, bf16_t* __restrict__ dst,
                               int M, int Kin, int Kout, int ldd) {
  int idx = blockIdx.x * blockDim.x + threadIdx.x;
  if (idx >= M * Kout) return;
  int m = idx / Kout, k = idx - m * Kout;
  float v = (k < Kin) ? src[(size_t)m * Kin + k] : 0.0f;
  dst[(size_t)m * ldd + k] = f2bf(v);
}

__global__ void bias_sum_kernel(const float* __restrict__ a, const float* __restrict__ b,
                                float* __restrict__ dst, int n) {
  int i = blockIdx.x * blockDim.x + threadIdx.x;
  if (i < n) dst[i] = a[i] + b[i];
}

// h0 (fp32) -> hbuf parity-0 (bf16), c0 -> cbuf (fp32)
__global__ void init_state_kernel(const float* __restrict__ h0, const float* __restrict__ c0,
                                  bf16_t* __restrict__ hbuf, float* __restrict__ cbuf) {
  int i = blockIdx.x * blockDim.x + threadIdx.x;
  if (i < B_ * H_) {
    hbuf[i] = f2bf(h0[i]);
    cbuf[i] = c0[i];
  }
}

__global__ void zero_bar_kernel(unsigned* bar) { *bar = 0u; }

// ---------- xg = Xchunk @ W_ih^T + (b_ih + b_hh) ----------
// Xin: [B, T, 512] bf16 (ld 512, only first Kdim cols valid)
// Wih: [2048, ldw] bf16   xg: [B, TC, 2048] fp32
// grid: (16 n-blocks of 128 cols, 128 b-blocks), 256 threads (8 waves)
__global__ void xg_gemm_kernel(const bf16_t* __restrict__ Xin,
                               const bf16_t* __restrict__ Wih,
                               const float* __restrict__ bsum,
                               float* __restrict__ xg,
                               int t0, int Kdim, int ldw) {
  int lane = threadIdx.x & 31, w = threadIdx.x >> 5;
  int b = blockIdx.y;
  int n0 = blockIdx.x * 128 + w * 16;
  const bf16_t* A = Xin + ((size_t)b * T_ + t0) * H_;

  v8f acc[8];
#pragma unroll
  for (int mt = 0; mt < 8; ++mt)
#pragma unroll
    for (int i = 0; i < 8; ++i) acc[mt][i] = 0.0f;

  for (int k = 0; k < Kdim; k += 32) {
    v16bf bfrag = load_frag(Wih, ldw, n0, k, lane);
#pragma unroll
    for (int mt = 0; mt < 8; ++mt) {
      v16bf afrag = load_frag(A, H_, mt * 16, k, lane);
      acc[mt] = WMMA_BF16(afrag, bfrag, acc[mt]);
    }
  }

  int n = n0 + (lane & 15);
  float bv = bsum[n];
#pragma unroll
  for (int mt = 0; mt < 8; ++mt)
#pragma unroll
    for (int v = 0; v < 8; ++v) {
      int tt = mt * 16 + v + 8 * (lane >> 4);
      xg[((size_t)b * TC_ + tt) * G4H + n] = acc[mt][v] + bv;
    }
}

// ---------- persistent recurrent scan over one T-chunk ----------
// 16 WGs x 256 threads. WG owns 32 hidden units (j0..j0+31) -> gate cols
// {g*512 + j0 .. +32} for g=0..3. Wave w: gate w>>1, unit-subtile w&1.
// Wave's W_hh slice (16 cols x 512 K = 16 KB = 16 fragments) is held in VGPRs
// for the whole chunk, so the per-step inner loop is pure ds_load + wmma.
// h double-buffered in global bf16; per-step device barrier (atomic + s_sleep).
// LDS (64 KB) is time-multiplexed per step:
//   phase 1: h staging, two 64KB halves of [128 x 256] bf16 (A operand via ds_load)
//   phase 2: gate staging [4][128][32] fp32
__global__ void lstm_scan_kernel(const float* __restrict__ xg,
                                 const bf16_t* __restrict__ Whh,   // [2048,512]
                                 bf16_t* __restrict__ hbuf,        // [2][B*H]
                                 float* __restrict__ cbuf,         // [B*H]
                                 bf16_t* __restrict__ ys,          // [B,T,H]
                                 float* __restrict__ outF,         // d_out
                                 unsigned* __restrict__ bar,
                                 int t0, int layer) {
  __shared__ __align__(16) char smem[64 * 1024];
  bf16_t* hsm = (bf16_t*)smem;   // [128][256] bf16 during GEMM phase
  float*  glf = (float*)smem;    // [4][128][32] fp32 during elementwise phase

  int tid = threadIdx.x;
  int lane = tid & 31, w = tid >> 5;
  int j0 = blockIdx.x * 32;
  int gate = w >> 1, jsub = w & 1;
  int nrow0 = gate * H_ + j0 + jsub * 16;   // W_hh row base (= gate column)

  // elementwise ownership: unit jj, batch rows mbase..mbase+15
  int jj = tid & 31;
  int mbase = (tid >> 5) * 16;

  // W_hh B-fragments resident in VGPRs for the entire chunk (16 x 8 VGPRs)
  v16bf bfr[16];
#pragma unroll
  for (int ks = 0; ks < 16; ++ks)
    bfr[ks] = load_frag(Whh, H_, nrow0, ks * 32, lane);

  float creg[16];
#pragma unroll
  for (int e = 0; e < 16; ++e)
    creg[e] = cbuf[(size_t)(mbase + e) * H_ + j0 + jj];

  for (int tt = 0; tt < TC_; ++tt) {
    int t = t0 + tt;
    const bf16_t* hcur = hbuf + (size_t)(t & 1) * (B_ * H_);
    bf16_t* hnxt = hbuf + (size_t)((t + 1) & 1) * (B_ * H_);

    // prefetch next step's xg gate lines so they overlap the GEMM
    if (tt + 1 < TC_) {
      int ep = jj & 15;
      size_t xp = ((size_t)(mbase + ep) * TC_ + (tt + 1)) * G4H + j0 + jj;
      __builtin_prefetch(&xg[xp], 0, 0);
      __builtin_prefetch(&xg[xp + 1 * H_], 0, 0);
      __builtin_prefetch(&xg[xp + 2 * H_], 0, 0);
      __builtin_prefetch(&xg[xp + 3 * H_], 0, 0);
    }

    // gates(recurrent part) = h @ Whh^T for this wave's 16 columns, all 128 rows
    v8f acc[8];
#pragma unroll
    for (int mt = 0; mt < 8; ++mt)
#pragma unroll
      for (int i = 0; i < 8; ++i) acc[mt][i] = 0.0f;

#pragma unroll
    for (int half = 0; half < 2; ++half) {
      // cooperative stage: hcur[:, half*256 .. +256) -> hsm[128][256]
      // 32768 elems as 4096 v8bf chunks, 16 per thread
#pragma unroll 4
      for (int it = 0; it < 16; ++it) {
        int cid = it * 256 + tid;
        int r = cid >> 5;            // 32 chunks per row
        int c8 = (cid & 31) << 3;    // element col
        *(v8bf*)(hsm + r * 256 + c8) =
            *(const v8bf*)(hcur + (size_t)r * H_ + half * 256 + c8);
      }
      __syncthreads();

#pragma unroll
      for (int ks = 0; ks < 8; ++ks) {
#pragma unroll
        for (int mt = 0; mt < 8; ++mt) {
          v16bf afrag = load_frag(hsm, 256, mt * 16, ks * 32, lane);
          acc[mt] = WMMA_BF16(afrag, bfr[half * 8 + ks], acc[mt]);
        }
      }
      __syncthreads();   // all LDS reads done before next stage / gate overwrite
    }

    // stage gates to LDS, gate-major (aliases h staging buffer)
#pragma unroll
    for (int mt = 0; mt < 8; ++mt)
#pragma unroll
      for (int v = 0; v < 8; ++v) {
        int m = mt * 16 + v + 8 * (lane >> 4);
        glf[((gate * B_) + m) * 32 + jsub * 16 + (lane & 15)] = acc[mt][v];
      }
    __syncthreads();

    // elementwise LSTM cell update for 16 (m, jj) pairs
#pragma unroll 4
    for (int e = 0; e < 16; ++e) {
      int m = mbase + e;
      size_t xr = ((size_t)m * TC_ + tt) * G4H + j0 + jj;
      float ig = glf[((0 * B_) + m) * 32 + jj] + xg[xr];
      float fg = glf[((1 * B_) + m) * 32 + jj] + xg[xr + 1 * H_];
      float gg = glf[((2 * B_) + m) * 32 + jj] + xg[xr + 2 * H_];
      float og = glf[((3 * B_) + m) * 32 + jj] + xg[xr + 3 * H_];
      float c = sigf(fg) * creg[e] + sigf(ig) * tanhf(gg);
      creg[e] = c;
      float h = sigf(og) * tanhf(c);
      bf16_t hb = f2bf(h);
      hnxt[(size_t)m * H_ + j0 + jj] = hb;
      ys[((size_t)m * T_ + t) * H_ + j0 + jj] = hb;
      if (t == T_ - 1) {
        size_t base = (size_t)B_ * T_ * V_;
        size_t idx = (size_t)m * H_ + j0 + jj;
        outF[base + (size_t)layer * B_ * H_ + idx] = h;          // hT
        outF[base + (size_t)(3 + layer) * B_ * H_ + idx] = c;    // cT
      }
    }

    // device-wide barrier: h writes must be visible before next step's GEMM
    __threadfence();
    __syncthreads();
    if (tid == 0) {
      __hip_atomic_fetch_add(bar, 1u, __ATOMIC_ACQ_REL, __HIP_MEMORY_SCOPE_AGENT);
      unsigned tgt = (unsigned)NWG_SCAN * (unsigned)(tt + 1);
      while (__hip_atomic_load(bar, __ATOMIC_ACQUIRE, __HIP_MEMORY_SCOPE_AGENT) < tgt)
        __builtin_amdgcn_s_sleep(4);
    }
    __syncthreads();
  }

  // persist c for next chunk
#pragma unroll
  for (int e = 0; e < 16; ++e)
    cbuf[(size_t)(mbase + e) * H_ + j0 + jj] = creg[e];
}

// ---------- logits = ys @ W_fc^T + b_fc ----------
// grid: 512 WGs x 256 threads; wave w -> rows row0+w*16, all 96 cols (6 tiles)
__global__ void logits_kernel(const bf16_t* __restrict__ ys,    // [B*T, 512]
                              const bf16_t* __restrict__ Wfc,   // [96, 512]
                              const float* __restrict__ bfc,    // [96]
                              float* __restrict__ out) {
  int lane = threadIdx.x & 31, w = threadIdx.x >> 5;
  int row0 = blockIdx.x * 128 + w * 16;

  v8f acc[6];
#pragma unroll
  for (int n = 0; n < 6; ++n)
#pragma unroll
    for (int i = 0; i < 8; ++i) acc[n][i] = 0.0f;

  for (int k = 0; k < H_; k += 32) {
    v16bf afrag = load_frag(ys, H_, row0, k, lane);
#pragma unroll
    for (int n = 0; n < 6; ++n) {
      v16bf bfrag = load_frag(Wfc, H_, n * 16, k, lane);
      acc[n] = WMMA_BF16(afrag, bfrag, acc[n]);
    }
  }

#pragma unroll
  for (int n = 0; n < 6; ++n)
#pragma unroll
    for (int v = 0; v < 8; ++v) {
      int m = row0 + v + 8 * (lane >> 4);
      int col = n * 16 + (lane & 15);
      out[(size_t)m * V_ + col] = acc[n][v] + bfc[col];
    }
}

// ---------- host launch ----------

extern "C" void kernel_launch(void* const* d_in, const int* in_sizes, int n_in,
                              void* d_out, int out_size, void* d_ws, size_t ws_size,
                              hipStream_t stream) {
  (void)in_sizes; (void)n_in; (void)out_size; (void)ws_size;

  const float* x  = (const float*)d_in[0];
  const float* h0 = (const float*)d_in[1];
  const float* c0 = (const float*)d_in[2];
  const float* W_ih[3] = {(const float*)d_in[3], (const float*)d_in[7],  (const float*)d_in[11]};
  const float* W_hh[3] = {(const float*)d_in[4], (const float*)d_in[8],  (const float*)d_in[12]};
  const float* b_ih[3] = {(const float*)d_in[5], (const float*)d_in[9],  (const float*)d_in[13]};
  const float* b_hh[3] = {(const float*)d_in[6], (const float*)d_in[10], (const float*)d_in[14]};
  const float* W_fc = (const float*)d_in[15];
  const float* b_fc = (const float*)d_in[16];
  float* out = (float*)d_out;

  char* ws = (char*)d_ws;
  size_t off = 0;
  auto alloc = [&](size_t bytes) -> char* {
    char* p = ws + off;
    off += (bytes + 255) & ~(size_t)255;
    return p;
  };

  float*  xg   = (float*) alloc((size_t)B_ * TC_ * G4H * 4);       // 128 MB
  bf16_t* ping = (bf16_t*)alloc((size_t)B_ * T_ * H_ * 2);         // 64 MB
  bf16_t* pong = (bf16_t*)alloc((size_t)B_ * T_ * H_ * 2);         // 64 MB
  bf16_t* wih[3];
  wih[0] = (bf16_t*)alloc((size_t)G4H * 128 * 2);
  wih[1] = (bf16_t*)alloc((size_t)G4H * H_ * 2);
  wih[2] = (bf16_t*)alloc((size_t)G4H * H_ * 2);
  bf16_t* whh[3];
  for (int l = 0; l < 3; ++l) whh[l] = (bf16_t*)alloc((size_t)G4H * H_ * 2);
  bf16_t* wfc  = (bf16_t*)alloc((size_t)V_ * H_ * 2);
  float*  bsum = (float*) alloc((size_t)3 * G4H * 4);
  bf16_t* hbuf = (bf16_t*)alloc((size_t)2 * B_ * H_ * 2);
  float*  cbuf = (float*) alloc((size_t)B_ * H_ * 4);
  unsigned* bar = (unsigned*)alloc(256);

  const int TPB = 256;
  auto blocks = [](size_t n) { return (unsigned)((n + 255) / 256); };

  // convert inputs/weights to bf16 (pad V=96 -> 128 for layer-0 K dim)
  cvt_pad_kernel<<<blocks((size_t)B_ * T_ * 128), TPB, 0, stream>>>(x, ping, B_ * T_, V_, 128, H_);
  cvt_pad_kernel<<<blocks((size_t)G4H * 128), TPB, 0, stream>>>(W_ih[0], wih[0], G4H, V_, 128, 128);
  cvt_pad_kernel<<<blocks((size_t)G4H * H_), TPB, 0, stream>>>(W_ih[1], wih[1], G4H, H_, H_, H_);
  cvt_pad_kernel<<<blocks((size_t)G4H * H_), TPB, 0, stream>>>(W_ih[2], wih[2], G4H, H_, H_, H_);
  for (int l = 0; l < 3; ++l) {
    cvt_pad_kernel<<<blocks((size_t)G4H * H_), TPB, 0, stream>>>(W_hh[l], whh[l], G4H, H_, H_, H_);
    bias_sum_kernel<<<blocks(G4H), TPB, 0, stream>>>(b_ih[l], b_hh[l], bsum + l * G4H, G4H);
  }
  cvt_pad_kernel<<<blocks((size_t)V_ * H_), TPB, 0, stream>>>(W_fc, wfc, V_, H_, H_, H_);

  bf16_t* bufs_in[3]  = {ping, pong, ping};
  bf16_t* bufs_out[3] = {pong, ping, pong};
  int Kdims[3] = {128, H_, H_};
  int ldws[3]  = {128, H_, H_};

  for (int l = 0; l < 3; ++l) {
    init_state_kernel<<<blocks((size_t)B_ * H_), TPB, 0, stream>>>(
        h0 + (size_t)l * B_ * H_, c0 + (size_t)l * B_ * H_, hbuf, cbuf);
    for (int ch = 0; ch < T_ / TC_; ++ch) {
      int t0 = ch * TC_;
      xg_gemm_kernel<<<dim3(G4H / 128, B_), TPB, 0, stream>>>(
          bufs_in[l], wih[l], bsum + l * G4H, xg, t0, Kdims[l], ldws[l]);
      zero_bar_kernel<<<1, 1, 0, stream>>>(bar);
      lstm_scan_kernel<<<NWG_SCAN, TPB, 0, stream>>>(
          xg, whh[l], hbuf, cbuf, bufs_out[l], out, bar, t0, l);
    }
  }

  logits_kernel<<<(B_ * T_) / 128, TPB, 0, stream>>>(pong, wfc, b_fc, out);
}
